// HyperbolicDistanceHead_38577396252949
// MI455X (gfx1250) — matrix operand order repaired and checked by reading, not compiled, not run on hardware
//
#include <hip/hip_runtime.h>

typedef _Float16 half_t;
typedef __attribute__((ext_vector_type(16))) _Float16 v16h;
typedef __attribute__((ext_vector_type(8)))  _Float16 v8h;
typedef __attribute__((ext_vector_type(8)))  float    v8f;
typedef __attribute__((ext_vector_type(4)))  float    v4f;

#define D_DIM   512
#define BM      128
#define BN      128
#define BK      32
#define KSTEPS  (D_DIM / BK)
#define LDSK    40            // padded halves per LDS tile row (80B, keeps 16B alignment)
#define EPSV    0.01f
#define MAXNORM 0.95f         // 0.96 - eps

// ---------------------------------------------------------------------------
// Kernel 1: Poincare-ball projection, f32 -> f16 + squared norm.
// One wave (32 lanes) per row of 512; each lane owns 16 contiguous floats.
// ---------------------------------------------------------------------------
__global__ __launch_bounds__(256) void project_rows_kernel(
    const float* __restrict__ src, half_t* __restrict__ dst,
    float* __restrict__ sq, const float* __restrict__ pre_scale, int rows)
{
  const int gwave = (blockIdx.x * 256 + threadIdx.x) >> 5;
  const int lane  = threadIdx.x & 31;
  if (gwave >= rows) return;

  const float s = pre_scale ? pre_scale[0] : 1.0f;
  const float* row = src + (size_t)gwave * D_DIM;

  float v[16];
  float ss = 0.0f;
#pragma unroll
  for (int i = 0; i < 4; ++i) {
    v4f t = __builtin_nontemporal_load((const v4f*)(row + lane * 16 + i * 4));
#pragma unroll
    for (int j = 0; j < 4; ++j) {
      float x = t[j] * s;
      v[i * 4 + j] = x;
      ss += x * x;
    }
  }
  // wave32 reduction
#pragma unroll
  for (int off = 16; off >= 1; off >>= 1)
    ss += __shfl_xor(ss, off);

  const float pscale = fminf(MAXNORM / (sqrtf(ss) + 1e-5f), 1.0f);

  v8h lo, hi;
#pragma unroll
  for (int i = 0; i < 8; ++i) {
    lo[i] = (half_t)(v[i]     * pscale);
    hi[i] = (half_t)(v[8 + i] * pscale);
  }
  half_t* drow = dst + (size_t)gwave * D_DIM + lane * 16;
  *(v8h*)(drow)     = lo;
  *(v8h*)(drow + 8) = hi;

  if (lane == 0) sq[gwave] = ss * pscale * pscale;   // post-projection ||.||^2
}

// ---------------------------------------------------------------------------
// Kernel 2: fused hyperbolic-distance head.
// C-tile 128x128 per workgroup (8 waves, each 32x64 = 2x4 WMMA accumulators).
// Staging uses CDNA5 async DMA (GLOBAL_LOAD_ASYNC_TO_LDS_B128, ASYNCcnt),
// double-buffered; epilogue computes -arccosh(arg)*logit_scale with NT stores.
// ---------------------------------------------------------------------------
__global__ __launch_bounds__(256) void hyper_head_kernel(
    const half_t* __restrict__ A,   // (N, 512) f16, projected hidden
    const half_t* __restrict__ B,   // (V, 512) f16, projected weight
    const float*  __restrict__ x2,  // (N)
    const float*  __restrict__ y2,  // (V)
    const float*  __restrict__ logit_scale_p,
    float* __restrict__ out, int N, int V)
{
  __shared__ alignas(16) half_t As[2][BM * LDSK];
  __shared__ alignas(16) half_t Bs[2][BN * LDSK];

  const int tid  = threadIdx.x;
  const int lane = tid & 31;
  const int wave = tid >> 5;
  const int m0 = (wave & 3) * 32;   // wave's row offset inside the 128-row tile
  const int n0 = (wave >> 2) * 64;  // wave's col offset inside the 128-col tile
  const int rowBase = blockIdx.y * BM;
  const int colBase = blockIdx.x * BN;

  // staging: each thread DMAs 2x16B chunks per matrix per K-step (async to LDS)
  const int sr = tid >> 2;          // 0..63
  const int sk = (tid & 3) * 8;     // 0,8,16,24 (halves)

  // LDS byte offsets (low 32 bits of the flat shared address = LDS offset)
  const unsigned ldsA0 = (unsigned)(uintptr_t)&As[0][sr * LDSK + sk];
  const unsigned ldsB0 = (unsigned)(uintptr_t)&Bs[0][sr * LDSK + sk];
  const unsigned ldsRowHop = 64u * LDSK * sizeof(half_t);       // +64 tile rows
  const unsigned bufHopA = (unsigned)(BM * LDSK * sizeof(half_t));
  const unsigned bufHopB = (unsigned)(BN * LDSK * sizeof(half_t));

  // global byte offsets (GVS form: 64-bit SGPR base + 32-bit VGPR offset)
  const unsigned gA0 = (unsigned)(((size_t)(rowBase + sr) * D_DIM + sk) * sizeof(half_t));
  const unsigned gB0 = (unsigned)(((size_t)(colBase + sr) * D_DIM + sk) * sizeof(half_t));
  const unsigned gRowHop = 64u * D_DIM * sizeof(half_t);        // +64 source rows
  const uint64_t baseA = (uint64_t)A;
  const uint64_t baseB = (uint64_t)B;

  auto issue_tile = [&](int buf, int kk) {
    const unsigned koff = (unsigned)(kk * BK * sizeof(half_t));
    const unsigned la = ldsA0 + (unsigned)buf * bufHopA;
    const unsigned lb = ldsB0 + (unsigned)buf * bufHopB;
    const unsigned ga = gA0 + koff;
    const unsigned gb = gB0 + koff;
    asm volatile("global_load_async_to_lds_b128 %0, %1, %2"
                 :: "v"(la), "v"(ga), "s"(baseA) : "memory");
    asm volatile("global_load_async_to_lds_b128 %0, %1, %2"
                 :: "v"(la + ldsRowHop), "v"(ga + gRowHop), "s"(baseA) : "memory");
    asm volatile("global_load_async_to_lds_b128 %0, %1, %2"
                 :: "v"(lb), "v"(gb), "s"(baseB) : "memory");
    asm volatile("global_load_async_to_lds_b128 %0, %1, %2"
                 :: "v"(lb + ldsRowHop), "v"(gb + gRowHop), "s"(baseB) : "memory");
  };

  // prologue: DMA K-step 0 into buffer 0
  issue_tile(0, 0);
  asm volatile("s_wait_asynccnt 0x0" ::: "memory");
  __syncthreads();

  v8f zero = {};
  v8f acc[2][4];
#pragma unroll
  for (int rt = 0; rt < 2; ++rt)
#pragma unroll
    for (int ct = 0; ct < 4; ++ct)
      acc[rt][ct] = zero;

  // ISA 7.12.2 fragment gather indices (wave32):
  // A (16x32 f16): lanes 0-15 row m, K {0..7,16..23}; lanes 16-31 row m, K {8..15,24..31}
  const int aRow = m0 + (lane & 15);
  const int kA0  = (lane >> 4) * 8;
  const int kA1  = 16 + kA0;
  // B (32x16 f16): lanes 0-15 col n, K {0..7,16..23}; lanes 16-31 col n, K {8..15,24..31}
  const int bCol = n0 + (lane & 15);
  const int kB   = (lane >> 4) * 8;

  for (int kk = 0; kk < KSTEPS; ++kk) {
    const int cur  = kk & 1;
    const bool more = (kk + 1) < KSTEPS;

    if (more) issue_tile(cur ^ 1, kk + 1);   // DMA next tile, overlaps WMMA

    v16h af[2], bf[4];
#pragma unroll
    for (int rt = 0; rt < 2; ++rt) {
      const half_t* ap = &As[cur][(aRow + rt * 16) * LDSK];
      v8h lo = *(const v8h*)(ap + kA0);
      v8h hi = *(const v8h*)(ap + kA1);
      af[rt] = __builtin_shufflevector(lo, hi, 0,1,2,3,4,5,6,7,8,9,10,11,12,13,14,15);
    }
#pragma unroll
    for (int ct = 0; ct < 4; ++ct) {
      const half_t* bp = &Bs[cur][(bCol + ct * 16) * LDSK + kB];
      v8h lo = *(const v8h*)(bp);
      v8h hi = *(const v8h*)(bp + 16);
      bf[ct] = __builtin_shufflevector(lo, hi, 0,1,2,3,4,5,6,7,8,9,10,11,12,13,14,15);
    }

#pragma unroll
    for (int rt = 0; rt < 2; ++rt)
#pragma unroll
      for (int ct = 0; ct < 4; ++ct)
        acc[rt][ct] = __builtin_amdgcn_wmma_f32_16x16x32_f16(
            false, af[rt], false, bf[ct], (short)0, acc[rt][ct], false, false);

    if (more) asm volatile("s_wait_asynccnt 0x0" ::: "memory");
    __syncthreads();
  }

  // ------------------ fused hyperbolic epilogue ------------------
  // C layout (f32 16x16): VGPR j, lanes 0-15 -> row j, N=lane; lanes 16-31 -> row 8+j
  const float ls = logit_scale_p[0];
  const int halfSel = (lane >> 4) * 8;
#pragma unroll
  for (int rt = 0; rt < 2; ++rt) {
    const int rbase = rowBase + m0 + rt * 16 + halfSel;
    float xx[8];
#pragma unroll
    for (int j = 0; j < 8; ++j) xx[j] = x2[rbase + j];
#pragma unroll
    for (int ct = 0; ct < 4; ++ct) {
      const int col = colBase + n0 + ct * 16 + (lane & 15);
      const float yy = y2[col];
      v8f c = acc[rt][ct];
#pragma unroll
      for (int j = 0; j < 8; ++j) {
        float d2    = fmaxf(xx[j] + yy - 2.0f * c[j], 0.0f);
        float denom = (1.0f - xx[j]) * (1.0f - yy);
        float arg   = 1.0f + 2.0f * d2 / (denom + EPSV);
        arg = fmaxf(arg, 1.0f + EPSV);
        float dist = __logf(arg + sqrtf(arg * arg - 1.0f));   // arccosh
        // output is written once, never re-read: stream past L2 (TH=NT)
        __builtin_nontemporal_store(-dist * ls, &out[(size_t)(rbase + j) * V + col]);
      }
    }
  }
}

// ---------------------------------------------------------------------------
extern "C" void kernel_launch(void* const* d_in, const int* in_sizes, int n_in,
                              void* d_out, int out_size, void* d_ws, size_t ws_size,
                              hipStream_t stream)
{
  const float* hs     = (const float*)d_in[0];   // (4,1024,512) f32
  const float* wt     = (const float*)d_in[1];   // (32000,512)  f32
  const float* hscale = (const float*)d_in[2];   // scalar
  const float* lscale = (const float*)d_in[3];   // scalar

  const int N = in_sizes[0] / D_DIM;   // 4096
  const int V = in_sizes[1] / D_DIM;   // 32000

  char* ws = (char*)d_ws;
  size_t off = 0;
  half_t* hA = (half_t*)(ws + off); off += (size_t)N * D_DIM * sizeof(half_t);
  off = (off + 255) & ~(size_t)255;
  half_t* wB = (half_t*)(ws + off); off += (size_t)V * D_DIM * sizeof(half_t);
  off = (off + 255) & ~(size_t)255;
  float* x2 = (float*)(ws + off);  off += (size_t)N * sizeof(float);
  off = (off + 255) & ~(size_t)255;
  float* y2 = (float*)(ws + off);

  project_rows_kernel<<<(N + 7) / 8, 256, 0, stream>>>(hs, hA, x2, hscale, N);
  project_rows_kernel<<<(V + 7) / 8, 256, 0, stream>>>(wt, wB, y2, nullptr, V);

  dim3 grid(V / BN, N / BM);   // 250 x 32
  hyper_head_kernel<<<grid, 256, 0, stream>>>(hA, wB, x2, y2, lscale,
                                              (float*)d_out, N, V);
}
